// CharWolfComponent_27530740367443
// MI455X (gfx1250) — matrix operand order repaired and checked by reading, not compiled, not run on hardware
//
#include <hip/hip_runtime.h>
#include <hip/hip_bf16.h>

typedef __attribute__((ext_vector_type(16))) _Float16 v16h;
typedef __attribute__((ext_vector_type(8)))  _Float16 v8h;
typedef __attribute__((ext_vector_type(8)))  float    v8f;

#define THREADS 256
#define ZROW 4288              // padded row stride of zxbcdt (4256 -> 4288 = 67*64)
#define AROW 40                // halves per LDS A-tile row (padded, 16B aligned, bank-spread)
#define BROW 40                // halves per LDS B-tile row

static inline int cdiv_ll(long long a, int b) { return (int)((a + (long long)b - 1) / b); }

// ---------------------------------------------------------------------------
// WMMA GEMM: C[M,N] = A[M,K] * W[K,N] (+ bias[N]); f32 in memory, f16 frags,
// f32 accumulate. Block = 256 thr = 8 waves -> 256x64 tile; wave = 64x32 tile
// (8 accumulators -> 8 v_wmma per K-step vs 12 ds_load_b128).
// LDS double-buffered, converted-once f16 tiles.
// Requires M%256==0, N%64==0, K%32==0.
// ---------------------------------------------------------------------------
__device__ inline void stage_tiles(const float* __restrict__ A, const float* __restrict__ W,
                                   int K, int N, int row0, int col0, int k0,
                                   _Float16* __restrict__ as, _Float16* __restrict__ bt, int t)
{
    // A tile: 256 rows x 32 k ; one row per thread: 8 float4 loads -> 4 v8h stores
    {
        const float4* g4 = (const float4*)(A + (size_t)(row0 + t) * K + k0);
        _Float16* dst = &as[t * AROW];
#pragma unroll
        for (int j = 0; j < 4; ++j) {
            float4 f0 = g4[2 * j], f1 = g4[2 * j + 1];
            v8h h;
            h[0]=(_Float16)f0.x; h[1]=(_Float16)f0.y; h[2]=(_Float16)f0.z; h[3]=(_Float16)f0.w;
            h[4]=(_Float16)f1.x; h[5]=(_Float16)f1.y; h[6]=(_Float16)f1.z; h[7]=(_Float16)f1.w;
            *(v8h*)(dst + 8 * j) = h;
        }
    }
    // B tile: 32 k x 64 c, stored transposed bt[c][k]; float4 loads along N
    {
        int k  = t >> 4;               // 0..15
        int c4 = (t & 15) << 2;        // 0,4,...,60
#pragma unroll
        for (int j = 0; j < 2; ++j) {
            int kk = k + 16 * j;
            float4 f = *(const float4*)(W + (size_t)(k0 + kk) * N + col0 + c4);
            bt[(c4 + 0) * BROW + kk] = (_Float16)f.x;
            bt[(c4 + 1) * BROW + kk] = (_Float16)f.y;
            bt[(c4 + 2) * BROW + kk] = (_Float16)f.z;
            bt[(c4 + 3) * BROW + kk] = (_Float16)f.w;
        }
    }
}

__device__ inline v16h frag_a(const _Float16* __restrict__ as, int row, int lane) {
    int kb = (lane >> 4) << 3;                    // 0 or 8
    const _Float16* p = as + (size_t)(row + (lane & 15)) * AROW + kb;
    v8h lo = *(const v8h*)p;                      // K kb..kb+7
    v8h hi = *(const v8h*)(p + 16);               // K kb+16..kb+23
    return __builtin_shufflevector(lo, hi, 0,1,2,3,4,5,6,7,8,9,10,11,12,13,14,15);
}

__device__ inline v16h frag_b(const _Float16* __restrict__ bt, int col, int lane) {
    int kb = (lane >> 4) << 3;                    // 0 or 8
    const _Float16* p = bt + (size_t)(col + (lane & 15)) * BROW + kb;
    v8h lo = *(const v8h*)p;                      // K kb..kb+7
    v8h hi = *(const v8h*)(p + 16);               // K kb+16..kb+23
    return __builtin_shufflevector(lo, hi, 0,1,2,3,4,5,6,7,8,9,10,11,12,13,14,15);
}

__global__ __launch_bounds__(THREADS) void gemm_f16w_kernel(
    const float* __restrict__ A, const float* __restrict__ W,
    const float* __restrict__ bias, float* __restrict__ C,
    int M, int K, int N)
{
    __shared__ _Float16 As[2][256 * AROW];        // 2 x 20480 B
    __shared__ _Float16 Bt[2][64 * BROW];         // 2 x  5120 B
    int t = threadIdx.x, lane = t & 31, wv = t >> 5;
    int row0 = blockIdx.y << 8;
    int col0 = blockIdx.x << 6;
    int wr = (wv >> 1) << 6;                      // wave row offset: 0,64,128,192
    int wc = (wv & 1) << 5;                       // wave col offset: 0,32

    v8f acc[4][2];
#pragma unroll
    for (int i = 0; i < 4; ++i)
#pragma unroll
        for (int j = 0; j < 2; ++j) acc[i][j] = (v8f){};

    stage_tiles(A, W, K, N, row0, col0, 0, As[0], Bt[0], t);
    int buf = 0;
    for (int k0 = 0; k0 < K; k0 += 32) {
        __syncthreads();
        if (k0 + 32 < K) {
            if (k0 + 64 < K)
                __builtin_prefetch(A + (size_t)(row0 + t) * K + k0 + 64, 0, 1);
            stage_tiles(A, W, K, N, row0, col0, k0 + 32, As[buf ^ 1], Bt[buf ^ 1], t);
        }
        v16h b0 = frag_b(Bt[buf], wc,      lane);
        v16h b1 = frag_b(Bt[buf], wc + 16, lane);
#pragma unroll
        for (int i = 0; i < 4; ++i) {
            v16h a = frag_a(As[buf], wr + 16 * i, lane);
            acc[i][0] = __builtin_amdgcn_wmma_f32_16x16x32_f16(false, a, false, b0, (short)0, acc[i][0], false, false);
            acc[i][1] = __builtin_amdgcn_wmma_f32_16x16x32_f16(false, a, false, b1, (short)0, acc[i][1], false, false);
        }
        buf ^= 1;
    }

    int crow = (lane >> 4) << 3;                  // 0 or 8
    int ccol = lane & 15;
    int gc = col0 + wc;
    float b0v = bias ? bias[gc + ccol]      : 0.f;
    float b1v = bias ? bias[gc + 16 + ccol] : 0.f;
#pragma unroll
    for (int i = 0; i < 4; ++i) {
        int gr = row0 + wr + 16 * i;
#pragma unroll
        for (int r = 0; r < 8; ++r) {
            size_t r0 = (size_t)(gr + crow + r) * N + gc + ccol;
            C[r0]      = acc[i][0][r] + b0v;
            C[r0 + 16] = acc[i][1][r] + b1v;
        }
    }
}

static void launch_gemm(const float* A, const float* W, const float* bias, float* C,
                        int M, int K, int N, hipStream_t s) {
    dim3 grid(N >> 6, M >> 8);
    gemm_f16w_kernel<<<grid, THREADS, 0, s>>>(A, W, bias, C, M, K, N);
}

// ---------------------------------------------------------------------------
// Elementwise / small kernels
// ---------------------------------------------------------------------------
__global__ void pad_w_kernel(const float* __restrict__ w, float* __restrict__ wp,
                             int N, int NP, long long total) {
    long long tid = (long long)blockIdx.x * THREADS + threadIdx.x;
    if (tid >= total) return;
    int c = (int)(tid % NP);
    long long k = tid / NP;
    wp[tid] = (c < N) ? w[k * N + c] : 0.f;
}

__global__ void ds_transpose_kernel(const float* __restrict__ w, float* __restrict__ W2) {
    int tid = blockIdx.x * THREADS + threadIdx.x;
    if (tid >= 2 * 1024 * 1024) return;
    int o = tid & 1023;
    int t = tid >> 10;
    int c = t & 1023;
    int k = t >> 10;                                  // 0..1
    W2[tid] = w[((size_t)o * 1024 + c) * 2 + k];      // W2[(k*1024+c)*1024+o] = w[o][c][k]
}

__global__ void conv_silu_kernel(const float* __restrict__ ZX, const float* __restrict__ cw,
                                 const float* __restrict__ cb, float* __restrict__ O,
                                 int L, long long total) {
    long long tid = (long long)blockIdx.x * THREADS + threadIdx.x;
    if (tid >= total) return;
    int c = (int)(tid % 2176);
    long long bl = tid / 2176;
    int l = (int)(bl % L);
    const float* base = ZX + (bl - l) * ZROW + 2048 + c;   // ZX[b, t=0, 2048+c]
    float s = cb[c];
#pragma unroll
    for (int j = 0; j < 4; ++j) {
        int t = l - 3 + j;
        if (t >= 0) s += base[(size_t)t * ZROW] * cw[c * 4 + j];
    }
    O[tid] = s / (1.f + expf(-s));                         // silu
}

__global__ void dt_kernel(const float* __restrict__ ZX, const float* __restrict__ dt_bias,
                          const float* __restrict__ A_log, float* __restrict__ DT,
                          float* __restrict__ DA, long long rows) {
    long long tid = (long long)blockIdx.x * THREADS + threadIdx.x;
    if (tid >= rows * 32) return;
    int h = (int)(tid & 31);
    long long bl = tid >> 5;
    float x  = ZX[bl * ZROW + 4224 + h] + dt_bias[h];
    float sp = (x > 20.f) ? x : log1pf(expf(x));           // softplus
    DT[tid] = sp;
    DA[tid] = expf(sp * (-expf(A_log[h])));
}

// one block per (b, head); 64 threads = headdim; 64-wide state in registers.
// B/C rows staged global->LDS with CDNA5 async-LDS loads (ASYNCcnt path).
__global__ __launch_bounds__(64) void ssm_scan_kernel(
    const float* __restrict__ CONV, const float* __restrict__ DT,
    const float* __restrict__ DA, const float* __restrict__ Dp,
    float* __restrict__ Y, int L)
{
    int b = blockIdx.x >> 5;
    int h = blockIdx.x & 31;
    int p = threadIdx.x;
    __shared__ float Bs[64], Cs[64], sc[2];
    float hreg[64];
#pragma unroll
    for (int n = 0; n < 64; ++n) hreg[n] = 0.f;
    float Dh = Dp[h];
    unsigned ldsB = (unsigned)(size_t)&Bs[p];
    unsigned ldsC = (unsigned)(size_t)&Cs[p];
    for (int t = 0; t < L; ++t) {
        size_t row = ((size_t)(b * L + t)) * 2176;
        const float* gB = CONV + row + 2048 + p;
        const float* gC = CONV + row + 2112 + p;
        asm volatile("global_load_async_to_lds_b32 %0, %1, off" :: "v"(ldsB), "v"(gB) : "memory");
        asm volatile("global_load_async_to_lds_b32 %0, %1, off" :: "v"(ldsC), "v"(gC) : "memory");
        if (p == 0) {
            long long di = ((long long)(b * L + t)) * 32 + h;
            sc[0] = DA[di]; sc[1] = DT[di];
        }
        asm volatile("s_wait_asynccnt 0x0" ::: "memory");
        __syncthreads();
        float xv  = CONV[row + h * 64 + p];
        float xdt = xv * sc[1];
        float da  = sc[0];
        float yv  = 0.f;
#pragma unroll
        for (int n = 0; n < 64; ++n) {
            hreg[n] = hreg[n] * da + xdt * Bs[n];
            yv += hreg[n] * Cs[n];
        }
        Y[((size_t)(b * L + t)) * 2048 + h * 64 + p] = yv + Dh * xv;
        __syncthreads();
    }
}

// O = rmsnorm(Y * silu(z)) * w, row width 2048; z = first 2048 cols of ZX row
__global__ __launch_bounds__(THREADS) void gated_rmsnorm_kernel(
    const float* __restrict__ Y, const float* __restrict__ ZX,
    const float* __restrict__ w, float* __restrict__ O)
{
    __shared__ float red[THREADS];
    size_t r = blockIdx.x;
    const float* y = Y + r * 2048;
    const float* z = ZX + r * (size_t)ZROW;
    float vals[8]; float s2 = 0.f;
#pragma unroll
    for (int j = 0; j < 8; ++j) {
        int i = threadIdx.x + j * THREADS;
        float zv = z[i];
        float v  = y[i] * (zv / (1.f + expf(-zv)));
        vals[j] = v; s2 += v * v;
    }
    red[threadIdx.x] = s2; __syncthreads();
    for (int st = THREADS / 2; st > 0; st >>= 1) {
        if (threadIdx.x < st) red[threadIdx.x] += red[threadIdx.x + st];
        __syncthreads();
    }
    float rs = rsqrtf(red[0] / 2048.f + 1e-5f);
#pragma unroll
    for (int j = 0; j < 8; ++j) {
        int i = threadIdx.x + j * THREADS;
        O[r * 2048 + i] = vals[j] * rs * w[i];
    }
}

// x = rmsnorm(x + m) * w, row width 1024 (in place on x)
__global__ __launch_bounds__(THREADS) void resid_rmsnorm_kernel(
    float* __restrict__ X, const float* __restrict__ Mo, const float* __restrict__ w)
{
    __shared__ float red[THREADS];
    size_t base = (size_t)blockIdx.x * 1024;
    float vals[4]; float s2 = 0.f;
#pragma unroll
    for (int j = 0; j < 4; ++j) {
        int i = threadIdx.x + j * THREADS;
        float v = X[base + i] + Mo[base + i];
        vals[j] = v; s2 += v * v;
    }
    red[threadIdx.x] = s2; __syncthreads();
    for (int st = THREADS / 2; st > 0; st >>= 1) {
        if (threadIdx.x < st) red[threadIdx.x] += red[threadIdx.x + st];
        __syncthreads();
    }
    float rs = rsqrtf(red[0] / 1024.f + 1e-5f);
#pragma unroll
    for (int j = 0; j < 4; ++j) {
        int i = threadIdx.x + j * THREADS;
        X[base + i] = vals[j] * rs * w[i];
    }
}

// layernorm row width 1024
__global__ __launch_bounds__(THREADS) void layernorm_kernel(
    const float* __restrict__ X, const float* __restrict__ g,
    const float* __restrict__ b, float* __restrict__ O)
{
    __shared__ float r1[THREADS], r2[THREADS];
    size_t base = (size_t)blockIdx.x * 1024;
    float vals[4]; float s = 0.f, s2 = 0.f;
#pragma unroll
    for (int j = 0; j < 4; ++j) {
        int i = threadIdx.x + j * THREADS;
        float v = X[base + i];
        vals[j] = v; s += v; s2 += v * v;
    }
    r1[threadIdx.x] = s; r2[threadIdx.x] = s2; __syncthreads();
    for (int st = THREADS / 2; st > 0; st >>= 1) {
        if (threadIdx.x < st) { r1[threadIdx.x] += r1[threadIdx.x + st]; r2[threadIdx.x] += r2[threadIdx.x + st]; }
        __syncthreads();
    }
    float mean = r1[0] / 1024.f;
    float var  = r2[0] / 1024.f - mean * mean;
    float rs   = rsqrtf(var + 1e-5f);
#pragma unroll
    for (int j = 0; j < 4; ++j) {
        int i = threadIdx.x + j * THREADS;
        O[base + i] = (vals[j] - mean) * rs * g[i] + b[i];
    }
}

__global__ void elu1_kernel(float* __restrict__ X, long long n) {
    long long tid = (long long)blockIdx.x * THREADS + threadIdx.x;
    if (tid >= n) return;
    float x = X[tid];
    X[tid] = x > 0.f ? x + 1.f : expf(x);                  // elu(x)+1
}

__global__ void gelu_kernel(float* __restrict__ X, long long n) {
    long long tid = (long long)blockIdx.x * THREADS + threadIdx.x;
    if (tid >= n) return;
    float x = X[tid];
    float u = 0.7978845608028654f * (x + 0.044715f * x * x * x);
    X[tid] = 0.5f * x * (1.f + tanhf(u));
}

__global__ void add_kernel(float* __restrict__ X, const float* __restrict__ Y, long long n) {
    long long tid = (long long)blockIdx.x * THREADS + threadIdx.x;
    if (tid >= n) return;
    X[tid] += Y[tid];
}

// kv[d][e] = sum_l k*v ; ksum[d] = sum_l k   (per b,head)
__global__ __launch_bounds__(THREADS) void kv_ksum_kernel(
    const float* __restrict__ K, const float* __restrict__ V,
    float* __restrict__ KV, float* __restrict__ KS, int Lc)
{
    int b = blockIdx.x >> 4, h = blockIdx.x & 15;
    int t = threadIdx.x;
    int d  = t >> 2;
    int e0 = (t & 3) << 4;
    __shared__ float ks[64], vs[64];
    float acc[16];
#pragma unroll
    for (int i = 0; i < 16; ++i) acc[i] = 0.f;
    float ksum = 0.f;
    for (int l = 0; l < Lc; ++l) {
        size_t row = ((size_t)(b * Lc + l)) * 1024 + h * 64;
        if (t < 64) ks[t] = K[row + t];
        else if (t < 128) vs[t - 64] = V[row + t - 64];
        __syncthreads();
        float kd = ks[d];
#pragma unroll
        for (int i = 0; i < 16; ++i) acc[i] += kd * vs[e0 + i];
        if ((t & 3) == 0) ksum += kd;
        __syncthreads();
    }
    size_t base = ((size_t)(b * 16 + h)) * 4096 + (size_t)d * 64 + e0;
#pragma unroll
    for (int i = 0; i < 16; ++i) KV[base + i] = acc[i];
    if ((t & 3) == 0) KS[(b * 16 + h) * 64 + d] = ksum;
}

// out[e] = (sum_d q_d * kv[d][e]) / (sum_d q_d * ksum_d + 1e-6)
__global__ __launch_bounds__(64) void attn_out_kernel(
    const float* __restrict__ Q, const float* __restrict__ KV,
    const float* __restrict__ KS, float* __restrict__ O, int Lc)
{
    int h  = blockIdx.x & 15;
    int bl = blockIdx.x >> 4;
    int e  = threadIdx.x;
    __shared__ float qs[64], red[64];
    size_t qrow = (size_t)bl * 1024 + h * 64;
    qs[e] = Q[qrow + e];
    __syncthreads();
    int b = bl / Lc;
    red[e] = qs[e] * KS[(b * 16 + h) * 64 + e];
    __syncthreads();
    for (int s = 32; s > 0; s >>= 1) { if (e < s) red[e] += red[e + s]; __syncthreads(); }
    float den = red[0] + 1e-6f;
    const float* kvp = KV + ((size_t)(b * 16 + h)) * 4096 + e;
    float num = 0.f;
#pragma unroll
    for (int d = 0; d < 64; ++d) num += qs[d] * kvp[(size_t)d * 64];
    O[qrow + e] = num / den;
}

// ---------------------------------------------------------------------------
// Host orchestration
// ---------------------------------------------------------------------------
struct MP {
    const float *in_proj, *conv_w, *conv_b, *dt_bias, *A_log, *D, *norm_w, *out_proj;
};

static void run_mamba(float* x, int L, const MP& p, const float* nw,
                      float* ZX, float* CONV, float* YSSM, float* DT, float* DA, float* WP,
                      hipStream_t s)
{
    const int rows = 4 * L;
    long long ptot = 1024LL * ZROW;
    pad_w_kernel<<<cdiv_ll(ptot, THREADS), THREADS, 0, s>>>(p.in_proj, WP, 4256, ZROW, ptot);
    launch_gemm(x, WP, nullptr, ZX, rows, 1024, ZROW, s);
    long long ctot = (long long)rows * 2176;
    conv_silu_kernel<<<cdiv_ll(ctot, THREADS), THREADS, 0, s>>>(ZX, p.conv_w, p.conv_b, CONV, L, ctot);
    long long dtot = (long long)rows * 32;
    dt_kernel<<<cdiv_ll(dtot, THREADS), THREADS, 0, s>>>(ZX, p.dt_bias, p.A_log, DT, DA, rows);
    ssm_scan_kernel<<<128, 64, 0, s>>>(CONV, DT, DA, p.D, YSSM, L);
    float* YN = CONV;                      // reuse conv buffer (2048 <= 2176 per row)
    gated_rmsnorm_kernel<<<rows, THREADS, 0, s>>>(YSSM, ZX, p.norm_w, YN);
    float* MOUT = YSSM;                    // reuse yssm buffer
    launch_gemm(YN, p.out_proj, nullptr, MOUT, rows, 2048, 1024, s);
    resid_rmsnorm_kernel<<<rows, THREADS, 0, s>>>(x, MOUT, nw);
}

extern "C" void kernel_launch(void* const* d_in, const int* in_sizes, int n_in,
                              void* d_out, int out_size, void* d_ws, size_t ws_size,
                              hipStream_t stream)
{
    const float* x_in = (const float*)d_in[0];
    const float* in_w = (const float*)d_in[1];
    const float* in_b = (const float*)d_in[2];
    MP m[3];
    for (int i = 0; i < 3; ++i) {
        int k = 3 + i * 8;
        m[i].in_proj  = (const float*)d_in[k + 0];
        m[i].conv_w   = (const float*)d_in[k + 1];
        m[i].conv_b   = (const float*)d_in[k + 2];
        m[i].dt_bias  = (const float*)d_in[k + 3];
        m[i].A_log    = (const float*)d_in[k + 4];
        m[i].D        = (const float*)d_in[k + 5];
        m[i].norm_w   = (const float*)d_in[k + 6];
        m[i].out_proj = (const float*)d_in[k + 7];
    }
    const float* n1    = (const float*)d_in[27];
    const float* n2    = (const float*)d_in[28];
    const float* n3    = (const float*)d_in[29];
    const float* ds1   = (const float*)d_in[30];
    const float* ds2   = (const float*)d_in[31];
    const float* wq    = (const float*)d_in[32];
    const float* wk    = (const float*)d_in[33];
    const float* wv    = (const float*)d_in[34];
    const float* wo    = (const float*)d_in[35];
    const float* ln1_g = (const float*)d_in[36];
    const float* ln1_b = (const float*)d_in[37];
    const float* ln2_g = (const float*)d_in[38];
    const float* ln2_b = (const float*)d_in[39];
    const float* ff1_w = (const float*)d_in[40];
    const float* ff1_b = (const float*)d_in[41];
    const float* ff2_w = (const float*)d_in[42];
    const float* ff2_b = (const float*)d_in[43];
    const float* out_g = (const float*)d_in[44];
    const float* out_b = (const float*)d_in[45];

    // workspace arena (float offsets)
    float* WS   = (float*)d_ws;
    float* X    = WS;                       //  8,388,608  (4*2048*1024)
    float* ZX   = X    +  8388608;          // 35,127,296  (8192*4288)
    float* CONV = ZX   + 35127296;          // 17,825,792  (8192*2176); reused as YNORM
    float* YSSM = CONV + 17825792;          // 16,777,216  (8192*2048); reused as MOUT
    float* DT   = YSSM + 16777216;          //    262,144
    float* DA   = DT   +   262144;          //    262,144
    float* DSW  = DA   +   262144;          //  2,097,152  (transposed ds weights)
    float* X2   = DSW  +  2097152;          //  4,194,304  (4*1024*1024)
    float* X3   = X2   +  4194304;          //  2,097,152  (4*512*1024)
    float* WP   = X3   +  2097152;          //  4,390,912  (padded in_proj 1024x4288)

    // attention scratch inside ZX region (free during attention stage)
    float* AQ   = ZX;                       // 2,097,152
    float* AK   = ZX +  2097152;
    float* AV   = ZX +  4194304;
    float* HLN  = ZX +  6291456;
    float* ATT  = ZX +  8388608;
    float* AOUT = ZX + 10485760;
    float* FF1  = ZX + 12582912;            // 8,388,608
    float* FF2  = ZX + 20971520;
    float* KV   = ZX + 23068672;            // 262,144
    float* KS   = ZX + 23330816;            // 4,096

    // 1. input projection: (8192,256) @ (256,1024) + b
    launch_gemm(x_in, in_w, in_b, X, 8192, 256, 1024, stream);

    // 2-3. mamba blocks at L=2048
    run_mamba(X, 2048, m[0], n1, ZX, CONV, YSSM, DT, DA, WP, stream);
    run_mamba(X, 2048, m[1], n2, ZX, CONV, YSSM, DT, DA, WP, stream);

    // 4. downsample 1: (4096, 2048) @ (2048, 1024)
    ds_transpose_kernel<<<cdiv_ll(2097152, THREADS), THREADS, 0, stream>>>(ds1, DSW);
    launch_gemm(X, DSW, nullptr, X2, 4096, 2048, 1024, stream);

    // 5. mamba block at L=1024
    run_mamba(X2, 1024, m[2], n3, ZX, CONV, YSSM, DT, DA, WP, stream);

    // 6. downsample 2: (2048, 2048) @ (2048, 1024)
    ds_transpose_kernel<<<cdiv_ll(2097152, THREADS), THREADS, 0, stream>>>(ds2, DSW);
    launch_gemm(X2, DSW, nullptr, X3, 2048, 2048, 1024, stream);

    // 7. linear attention block (Lc=512, rows=2048)
    layernorm_kernel<<<2048, THREADS, 0, stream>>>(X3, ln1_g, ln1_b, HLN);
    launch_gemm(HLN, wq, nullptr, AQ, 2048, 1024, 1024, stream);
    launch_gemm(HLN, wk, nullptr, AK, 2048, 1024, 1024, stream);
    launch_gemm(HLN, wv, nullptr, AV, 2048, 1024, 1024, stream);
    elu1_kernel<<<cdiv_ll(2097152, THREADS), THREADS, 0, stream>>>(AQ, 2097152);
    elu1_kernel<<<cdiv_ll(2097152, THREADS), THREADS, 0, stream>>>(AK, 2097152);
    kv_ksum_kernel<<<64, THREADS, 0, stream>>>(AK, AV, KV, KS, 512);
    attn_out_kernel<<<32768, 64, 0, stream>>>(AQ, KV, KS, ATT, 512);
    launch_gemm(ATT, wo, nullptr, AOUT, 2048, 1024, 1024, stream);
    add_kernel<<<cdiv_ll(2097152, THREADS), THREADS, 0, stream>>>(X3, AOUT, 2097152);
    layernorm_kernel<<<2048, THREADS, 0, stream>>>(X3, ln2_g, ln2_b, HLN);
    launch_gemm(HLN, ff1_w, ff1_b, FF1, 2048, 1024, 4096, stream);
    gelu_kernel<<<cdiv_ll(8388608, THREADS), THREADS, 0, stream>>>(FF1, 8388608);
    launch_gemm(FF1, ff2_w, ff2_b, FF2, 2048, 4096, 1024, stream);
    add_kernel<<<cdiv_ll(2097152, THREADS), THREADS, 0, stream>>>(X3, FF2, 2097152);

    // 8. final layernorm -> output
    layernorm_kernel<<<2048, THREADS, 0, stream>>>(X3, out_g, out_b, (float*)d_out);
}
